// NRFormer_22041772163543
// MI455X (gfx1250) — compile-verified, matrix operand order checked by reading, not run on hardware
//
#include <hip/hip_runtime.h>
#include <hip/hip_bf16.h>
#include <math.h>

typedef __attribute__((ext_vector_type(16))) __bf16 v16bf;
typedef __attribute__((ext_vector_type(8)))  float  v8f;
typedef __attribute__((ext_vector_type(4)))  unsigned int u32x4;
typedef __attribute__((ext_vector_type(8)))  int i32x8;
typedef __attribute__((ext_vector_type(4)))  int i32x4;

#define BATCH   16
#define NSEQ    512
#define DMODEL  256
#define NHEAD   8
#define DKH     32
#define DFF     1024
#define NLAYER  4
#define MTOK    (BATCH * NSEQ)   // 8192 tokens

// ---------------------------------------------------------------------------
// Gather one WMMA 16-wide bf16 operand row per lane (ISA 7.12.2 layout):
// lane (0-15 / 16-31) holds row M = lane&15; VGPR pairs hold
//   K = {lg*8+0..7} then {16+lg*8+0..7}.  B-operand is symmetric over W^T rows.
// ---------------------------------------------------------------------------
__device__ __forceinline__ v16bf load_op16(const __bf16* rowbase, int k0, int lg) {
    union { v16bf v; unsigned int u[8]; } r;
    const unsigned int* p0 = (const unsigned int*)(rowbase + k0 + lg * 8);
    const unsigned int* p1 = (const unsigned int*)(rowbase + k0 + 16 + lg * 8);
#pragma unroll
    for (int i = 0; i < 4; ++i) { r.u[i] = p0[i]; r.u[4 + i] = p1[i]; }
    return r.v;
}

// ---------------------------------------------------------------------------
// Tensor Data Mover: issue a 2D tile load (16 rows x K bf16) into LDS.
// D# built per CDNA5 ISA §8: group0 = {count, lds_addr, global_addr, type=2},
// group1 = {data_size=2B, tensor dims/strides, tile 16 x K}.
// ---------------------------------------------------------------------------
#if defined(__has_builtin)
#if __has_builtin(__builtin_amdgcn_tensor_load_to_lds)
#define HAVE_TDM 1
#endif
#endif

__device__ __forceinline__ void tdm_load_tile_16xK(unsigned long long gaddr,
                                                   unsigned int lds_addr,
                                                   unsigned int K,
                                                   unsigned int Ncols) {
#ifdef HAVE_TDM
    u32x4 g0;
    g0[0] = 1u;                                  // count=1, is_restore=0
    g0[1] = lds_addr;                            // lds_addr[31:0]
    g0[2] = (unsigned int)gaddr;                 // global_addr[31:0]
    g0[3] = (unsigned int)((gaddr >> 32) & 0x1FFFFFFull) | (2u << 30); // addr[56:32] | type=2
    i32x8 g1;
    g1[0] = (int)(1u << 16);                     // wg_mask=0, data_size=1 (2 bytes)
    g1[1] = (int)((K & 0xFFFFu) << 16);          // tensor_dim0[15:0] @ bits 63:48
    g1[2] = (int)(((K >> 16) & 0xFFFFu) | ((Ncols & 0xFFFFu) << 16)); // dim0 hi | dim1 lo
    g1[3] = (int)(((Ncols >> 16) & 0xFFFFu) | ((K & 0xFFFFu) << 16)); // dim1 hi | tile_dim0=K
    g1[4] = (int)16;                             // tile_dim1=16 rows, tile_dim2=0
    g1[5] = (int)K;                              // tensor_dim0_stride[31:0]
    g1[6] = 0;                                   // stride0 hi | dim1_stride lo
    g1[7] = 0;
    i32x4 z4 = {0, 0, 0, 0};
#if __clang_major__ >= 23
    i32x8 z8 = {0, 0, 0, 0, 0, 0, 0, 0};
    __builtin_amdgcn_tensor_load_to_lds(g0, g1, z4, z4, z8, 0);
#else
    __builtin_amdgcn_tensor_load_to_lds(g0, g1, z4, z4, 0);
#endif
#else
    (void)gaddr; (void)lds_addr; (void)K; (void)Ncols;
#endif
}

// ---------------------------------------------------------------------------
// x = input + pos ; xv = input_v + pos  (fp32 residual stream + bf16 copies)
// ---------------------------------------------------------------------------
__global__ void prep_kernel(const float* __restrict__ input,
                            const float* __restrict__ input_v,
                            const float* __restrict__ pos,
                            float* __restrict__ xF,
                            __bf16* __restrict__ xB,
                            __bf16* __restrict__ xvB) {
    int idx = blockIdx.x * blockDim.x + threadIdx.x;
    if (idx >= MTOK * DMODEL) return;
    int dd = idx & (DMODEL - 1);
    int nn = (idx >> 8) & (NSEQ - 1);
    float p = pos[nn * DMODEL + dd];
    float a = input[idx] + p;
    float b = input_v[idx] + p;
    xF[idx] = a;
    xB[idx] = (__bf16)a;
    xvB[idx] = (__bf16)b;
}

// f32 [K x N] row-major  ->  bf16 [N x K] (transposed) for B-operand gathers
__global__ void convT_kernel(const float* __restrict__ W, __bf16* __restrict__ Wt,
                             int K, int N) {
    int idx = blockIdx.x * blockDim.x + threadIdx.x;
    if (idx >= K * N) return;
    int k = idx / N, n = idx - k * N;
    Wt[(size_t)n * K + k] = (__bf16)W[idx];
}

// ---------------------------------------------------------------------------
// Register-blocked WMMA GEMM: each wave computes a 32x32 C tile (2x2 WMMA
// fragments, 4 WMMAs per K-step of 32) => 2x fragment reuse vs 16x16 tiling.
// C[M x Ncols] = A_bf16[M x K] @ Wt_bf16[Ncols x K]^T + bias, with epilogues.
// ---------------------------------------------------------------------------
enum { MODE_F32R = 0, MODE_BF16 = 1, MODE_GELU = 2, MODE_VT = 3 };

template <int MODE>
__global__ __launch_bounds__(128) void gemm_bf16_wmma(
    const __bf16* __restrict__ A, int lda,
    const __bf16* __restrict__ Wt,
    const float* __restrict__ bias,
    int M, int Ncols, int K,
    float* __restrict__ outF, const float* __restrict__ resid, int ldf,
    __bf16* __restrict__ outB, int ldb) {
    const int lane = threadIdx.x & 31;
    const int wave = threadIdx.x >> 5;
    const int lg = lane >> 4;
    const int nl = lane & 15;
    const int tilesN = Ncols >> 5;             // 32-wide column blocks
    int tile = blockIdx.x * 4 + wave;          // wave-uniform
    int tm = tile / tilesN;
    int tn = tile - tm * tilesN;
    if (tm * 32 >= M) return;                  // wave-uniform guard (EXEC stays all-ones)

    const __bf16* arow0 = A + (size_t)(tm * 32 + nl) * lda;
    const __bf16* arow1 = arow0 + (size_t)16 * lda;
    const __bf16* wrow0 = Wt + (size_t)(tn * 32 + nl) * K;
    const __bf16* wrow1 = wrow0 + (size_t)16 * K;
    v8f acc00 = {0.f, 0.f, 0.f, 0.f, 0.f, 0.f, 0.f, 0.f};
    v8f acc01 = acc00, acc10 = acc00, acc11 = acc00;
    for (int k = 0; k < K; k += 32) {
        if (k + 32 < K) {
            __builtin_prefetch(arow0 + k + 32, 0, 1);   // global_prefetch_b8
            __builtin_prefetch(wrow0 + k + 32, 0, 1);
        }
        v16bf a0 = load_op16(arow0, k, lg);
        v16bf a1 = load_op16(arow1, k, lg);
        v16bf b0 = load_op16(wrow0, k, lg);
        v16bf b1 = load_op16(wrow1, k, lg);
        acc00 = __builtin_amdgcn_wmma_f32_16x16x32_bf16(false, a0, false, b0, (short)0, acc00, false, false);
        acc01 = __builtin_amdgcn_wmma_f32_16x16x32_bf16(false, a0, false, b1, (short)0, acc01, false, false);
        acc10 = __builtin_amdgcn_wmma_f32_16x16x32_bf16(false, a1, false, b0, (short)0, acc10, false, false);
        acc11 = __builtin_amdgcn_wmma_f32_16x16x32_bf16(false, a1, false, b1, (short)0, acc11, false, false);
    }
#pragma unroll
    for (int j = 0; j < 8; ++j) {
        int m0 = tm * 32 + j + 8 * lg;         // C layout: VGPR j -> row j (+8 for lanes 16-31)
        int m1 = m0 + 16;
        int n0 = tn * 32 + nl;
        int n1 = n0 + 16;
        float v00 = acc00[j] + bias[n0];
        float v01 = acc01[j] + bias[n1];
        float v10 = acc10[j] + bias[n0];
        float v11 = acc11[j] + bias[n1];
        if (MODE == MODE_F32R) {
            outF[(size_t)m0 * ldf + n0] = v00 + resid[(size_t)m0 * ldf + n0];
            outF[(size_t)m0 * ldf + n1] = v01 + resid[(size_t)m0 * ldf + n1];
            outF[(size_t)m1 * ldf + n0] = v10 + resid[(size_t)m1 * ldf + n0];
            outF[(size_t)m1 * ldf + n1] = v11 + resid[(size_t)m1 * ldf + n1];
        } else if (MODE == MODE_BF16) {
            outB[(size_t)m0 * ldb + n0] = (__bf16)v00;
            outB[(size_t)m0 * ldb + n1] = (__bf16)v01;
            outB[(size_t)m1 * ldb + n0] = (__bf16)v10;
            outB[(size_t)m1 * ldb + n1] = (__bf16)v11;
        } else if (MODE == MODE_GELU) {
            outB[(size_t)m0 * ldb + n0] = (__bf16)(0.5f * v00 * (1.0f + erff(v00 * 0.70710678118654752f)));
            outB[(size_t)m0 * ldb + n1] = (__bf16)(0.5f * v01 * (1.0f + erff(v01 * 0.70710678118654752f)));
            outB[(size_t)m1 * ldb + n0] = (__bf16)(0.5f * v10 * (1.0f + erff(v10 * 0.70710678118654752f)));
            outB[(size_t)m1 * ldb + n1] = (__bf16)(0.5f * v11 * (1.0f + erff(v11 * 0.70710678118654752f)));
        } else { // MODE_VT: store V head-transposed: vt[b][h][dk][token]
            int b0_ = m0 >> 9, t0 = m0 & (NSEQ - 1);
            int b1_ = m1 >> 9, t1 = m1 & (NSEQ - 1);
            int h0 = n0 >> 5, d0 = n0 & 31;
            int h1 = n1 >> 5, d1 = n1 & 31;
            outB[(((size_t)b0_ * NHEAD + h0) * DKH + d0) * NSEQ + t0] = (__bf16)v00;
            outB[(((size_t)b0_ * NHEAD + h1) * DKH + d1) * NSEQ + t0] = (__bf16)v01;
            outB[(((size_t)b1_ * NHEAD + h0) * DKH + d0) * NSEQ + t1] = (__bf16)v10;
            outB[(((size_t)b1_ * NHEAD + h1) * DKH + d1) * NSEQ + t1] = (__bf16)v11;
        }
    }
}

// ---------------------------------------------------------------------------
// Wo projection with residual, using the Tensor Data Mover: each wave DMAs its
// 16x256 B tile (rows of Wo^T) into LDS (tensor_load_to_lds), preloads its A
// fragments from global while the DMA is in flight, then s_wait_tensorcnt and
// runs the K-loop gathering the B operand from LDS.
// ---------------------------------------------------------------------------
__global__ __launch_bounds__(128) void gemm_wo_tdm(
    const __bf16* __restrict__ A,            // [MTOK x 256]
    const __bf16* __restrict__ Wt,           // [256 x 256] (Wo^T)
    const float* __restrict__ bias,
    float* __restrict__ outF, const float* __restrict__ resid) {
    __shared__ __bf16 wtile[4][16 * DMODEL]; // 32 KB: one 16x256 tile per wave
    const int lane = threadIdx.x & 31;
    const int wave = threadIdx.x >> 5;
    const int lg = lane >> 4;
    const int nl = lane & 15;
    const int tilesN = DMODEL / 16;          // 16
    int tile = blockIdx.x * 4 + wave;
    int tm = tile / tilesN;
    int tn = tile - tm * tilesN;

#ifdef HAVE_TDM
    unsigned long long gaddr = (unsigned long long)(uintptr_t)(Wt + (size_t)tn * 16 * DMODEL);
    unsigned int laddr = (unsigned int)(uintptr_t)(&wtile[wave][0]);
    tdm_load_tile_16xK(gaddr, laddr, DMODEL, DMODEL);
#endif

    const __bf16* arow = A + (size_t)(tm * 16 + nl) * DMODEL;
    v16bf afr[8];
#pragma unroll
    for (int kk = 0; kk < 8; ++kk) afr[kk] = load_op16(arow, kk * 32, lg);

#ifdef HAVE_TDM
    __builtin_amdgcn_s_wait_tensorcnt(0);
    const __bf16* wrow = &wtile[wave][nl * DMODEL];
#else
    const __bf16* wrow = Wt + (size_t)(tn * 16 + nl) * DMODEL;
#endif

    v8f acc = {0.f, 0.f, 0.f, 0.f, 0.f, 0.f, 0.f, 0.f};
#pragma unroll
    for (int kk = 0; kk < 8; ++kk) {
        v16bf b = load_op16(wrow, kk * 32, lg);
        acc = __builtin_amdgcn_wmma_f32_16x16x32_bf16(false, afr[kk], false, b, (short)0, acc, false, false);
    }
#pragma unroll
    for (int j = 0; j < 8; ++j) {
        int m = tm * 16 + j + 8 * lg;
        int nn = tn * 16 + nl;
        float v = acc[j] + bias[nn];
        outF[(size_t)m * DMODEL + nn] = v + resid[(size_t)m * DMODEL + nn];
    }
}

// ---------------------------------------------------------------------------
// Attention: one wave per (batch, head, 16-query tile).  Branchless mask via
// template; scores in LDS, two-lane-per-row softmax, att@V via WMMA.
// ---------------------------------------------------------------------------
template <int USEMASK>
__global__ __launch_bounds__(32) void attn_kernel(
    const __bf16* __restrict__ Q, const __bf16* __restrict__ Kmat,
    const __bf16* __restrict__ Vt, const int* __restrict__ mask,
    __bf16* __restrict__ O) {
    __shared__ float  sc[16 * NSEQ];   // 32 KB scores
    __shared__ __bf16 pb[16 * NSEQ];   // 16 KB probabilities (bf16)

    int bid = blockIdx.x;
    int qt = bid & 31;
    int h  = (bid >> 5) & 7;
    int b  = bid >> 8;
    int lane = threadIdx.x;
    int lg = lane >> 4, nl = lane & 15;
    int q0 = qt * 16;

    const __bf16* qrow = Q + (size_t)(b * NSEQ + q0 + nl) * DMODEL + h * DKH;
    v16bf qa = load_op16(qrow, 0, lg);
    const float scale = 0.17677669529663687f;  // 1/sqrt(32)

    for (int kt = 0; kt < NSEQ / 16; ++kt) {
        const __bf16* krow = Kmat + (size_t)(b * NSEQ + kt * 16 + nl) * DMODEL + h * DKH;
        v16bf kb = load_op16(krow, 0, lg);
        v8f c = {0.f, 0.f, 0.f, 0.f, 0.f, 0.f, 0.f, 0.f};
        c = __builtin_amdgcn_wmma_f32_16x16x32_bf16(false, qa, false, kb, (short)0, c, false, false);
#pragma unroll
        for (int j = 0; j < 8; ++j) {
            int m = j + 8 * lg;
            float s = c[j] * scale;
            if (USEMASK) {
                int mv = mask[(size_t)(q0 + m) * NSEQ + kt * 16 + nl];
                s = mv ? -3.0e38f : s;         // branchless v_cndmask
            }
            sc[m * NSEQ + kt * 16 + nl] = s;
        }
    }
    __syncthreads();

    {   // softmax: lane pair (nl, nl+16) handles row nl, halves split by lg
        int base = nl * NSEQ + lg * 256;
        float mx = -3.4e38f;
        for (int i = 0; i < 256; ++i) mx = fmaxf(mx, sc[base + i]);
        mx = fmaxf(mx, __shfl_xor(mx, 16, 32));
        float sum = 0.f;
        for (int i = 0; i < 256; ++i) {
            float e = __expf(sc[base + i] - mx);
            sc[base + i] = e;
            sum += e;
        }
        sum += __shfl_xor(sum, 16, 32);
        float inv = 1.0f / sum;
        for (int i = 0; i < 256; ++i) pb[base + i] = (__bf16)(sc[base + i] * inv);
    }
    __syncthreads();

    v8f acc0 = {0.f, 0.f, 0.f, 0.f, 0.f, 0.f, 0.f, 0.f};
    v8f acc1 = acc0;
    const __bf16* prow = pb + nl * NSEQ;                                  // A row m = nl
    const __bf16* v0 = Vt + ((size_t)(b * NHEAD + h) * DKH + nl) * NSEQ;  // B^T rows
    const __bf16* v1 = v0 + 16 * NSEQ;
    for (int kk = 0; kk < NSEQ; kk += 32) {
        v16bf aa  = load_op16(prow, kk, lg);
        v16bf vb0 = load_op16(v0, kk, lg);
        v16bf vb1 = load_op16(v1, kk, lg);
        acc0 = __builtin_amdgcn_wmma_f32_16x16x32_bf16(false, aa, false, vb0, (short)0, acc0, false, false);
        acc1 = __builtin_amdgcn_wmma_f32_16x16x32_bf16(false, aa, false, vb1, (short)0, acc1, false, false);
    }
#pragma unroll
    for (int j = 0; j < 8; ++j) {
        int m = j + 8 * lg;
        size_t orow = (size_t)(b * NSEQ + q0 + m) * DMODEL + h * DKH;
        O[orow + nl]      = (__bf16)acc0[j];
        O[orow + 16 + nl] = (__bf16)acc1[j];
    }
}

// ---------------------------------------------------------------------------
// LayerNorm over D=256; one wave per token, 8 elements per lane.
// ---------------------------------------------------------------------------
__global__ __launch_bounds__(128) void layernorm_kernel(
    const float* __restrict__ in, const float* __restrict__ gamma,
    const float* __restrict__ beta, float* __restrict__ outF,
    __bf16* __restrict__ outB) {
    int tok = blockIdx.x * 4 + (threadIdx.x >> 5);
    int lane = threadIdx.x & 31;
    const float* row = in + (size_t)tok * DMODEL;
    float vals[8];
    float s = 0.f;
#pragma unroll
    for (int i = 0; i < 8; ++i) { vals[i] = row[lane * 8 + i]; s += vals[i]; }
#pragma unroll
    for (int off = 16; off; off >>= 1) s += __shfl_xor(s, off, 32);
    float mean = s * (1.0f / DMODEL);
    float vs = 0.f;
#pragma unroll
    for (int i = 0; i < 8; ++i) { float d = vals[i] - mean; vs += d * d; }
#pragma unroll
    for (int off = 16; off; off >>= 1) vs += __shfl_xor(vs, off, 32);
    float inv = rsqrtf(vs * (1.0f / DMODEL) + 1e-5f);
#pragma unroll
    for (int i = 0; i < 8; ++i) {
        int c = lane * 8 + i;
        float o = (vals[i] - mean) * inv * gamma[c] + beta[c];
        outF[(size_t)tok * DMODEL + c] = o;
        if (outB) outB[(size_t)tok * DMODEL + c] = (__bf16)o;
    }
}

// ---------------------------------------------------------------------------
extern "C" void kernel_launch(void* const* d_in, const int* in_sizes, int n_in,
                              void* d_out, int out_size, void* d_ws, size_t ws_size,
                              hipStream_t stream) {
    (void)in_sizes; (void)n_in; (void)out_size; (void)ws_size;
    const float* input   = (const float*)d_in[0];
    const float* input_v = (const float*)d_in[1];
    const float* pos     = (const float*)d_in[2];
    const float* Wq = (const float*)d_in[3];   const float* bq  = (const float*)d_in[4];
    const float* Wk = (const float*)d_in[5];   const float* bk  = (const float*)d_in[6];
    const float* Wv = (const float*)d_in[7];   const float* bv  = (const float*)d_in[8];
    const float* Wo = (const float*)d_in[9];   const float* bo  = (const float*)d_in[10];
    const float* W1 = (const float*)d_in[11];  const float* b1  = (const float*)d_in[12];
    const float* W2 = (const float*)d_in[13];  const float* b2  = (const float*)d_in[14];
    const float* g1 = (const float*)d_in[15];  const float* be1 = (const float*)d_in[16];
    const float* g2 = (const float*)d_in[17];  const float* be2 = (const float*)d_in[18];
    const float* gf = (const float*)d_in[19];  const float* bf_ = (const float*)d_in[20];
    const int*  mask = (const int*)d_in[21];

    // ---- workspace carve-up (~61 MB) ----
    char* wsp = (char*)d_ws;
    auto alloc = [&](size_t bytes) -> char* {
        char* p = wsp;
        wsp += (bytes + 255) & ~(size_t)255;
        return p;
    };
    float*  xF   = (float*)alloc((size_t)MTOK * DMODEL * 4);
    float*  tmp  = (float*)alloc((size_t)MTOK * DMODEL * 4);
    __bf16* xB   = (__bf16*)alloc((size_t)MTOK * DMODEL * 2);
    __bf16* xvB  = (__bf16*)alloc((size_t)MTOK * DMODEL * 2);
    __bf16* qB   = (__bf16*)alloc((size_t)MTOK * DMODEL * 2);
    __bf16* kB   = (__bf16*)alloc((size_t)MTOK * DMODEL * 2);
    __bf16* vtB  = (__bf16*)alloc((size_t)MTOK * DMODEL * 2);
    __bf16* oB   = (__bf16*)alloc((size_t)MTOK * DMODEL * 2);
    __bf16* h1B  = (__bf16*)alloc((size_t)MTOK * DFF * 2);
    __bf16* WqT  = (__bf16*)alloc((size_t)NLAYER * DMODEL * DMODEL * 2);
    __bf16* WkT  = (__bf16*)alloc((size_t)NLAYER * DMODEL * DMODEL * 2);
    __bf16* WvT  = (__bf16*)alloc((size_t)NLAYER * DMODEL * DMODEL * 2);
    __bf16* WoT  = (__bf16*)alloc((size_t)NLAYER * DMODEL * DMODEL * 2);
    __bf16* W1T  = (__bf16*)alloc((size_t)NLAYER * DMODEL * DFF * 2);
    __bf16* W2T  = (__bf16*)alloc((size_t)NLAYER * 512 * 128 * 2);

    // ---- weight conversion (f32 -> transposed bf16) ----
    const int eD = DMODEL * DMODEL;       // 65536
    const int e1 = DMODEL * DFF;          // 262144
    const int e2 = 512 * 128;             // 65536
    for (int l = 0; l < NLAYER; ++l) {
        convT_kernel<<<(eD + 255) / 256, 256, 0, stream>>>(Wq + (size_t)l * eD, WqT + (size_t)l * eD, DMODEL, DMODEL);
        convT_kernel<<<(eD + 255) / 256, 256, 0, stream>>>(Wk + (size_t)l * eD, WkT + (size_t)l * eD, DMODEL, DMODEL);
        convT_kernel<<<(eD + 255) / 256, 256, 0, stream>>>(Wv + (size_t)l * eD, WvT + (size_t)l * eD, DMODEL, DMODEL);
        convT_kernel<<<(eD + 255) / 256, 256, 0, stream>>>(Wo + (size_t)l * eD, WoT + (size_t)l * eD, DMODEL, DMODEL);
        convT_kernel<<<(e1 + 255) / 256, 256, 0, stream>>>(W1 + (size_t)l * e1, W1T + (size_t)l * e1, DMODEL, DFF);
        convT_kernel<<<(e2 + 255) / 256, 256, 0, stream>>>(W2 + (size_t)l * e2, W2T + (size_t)l * e2, 512, 128);
    }

    prep_kernel<<<(MTOK * DMODEL + 255) / 256, 256, 0, stream>>>(input, input_v, pos, xF, xB, xvB);

    const int blkD   = ((MTOK / 32) * (DMODEL / 32)) / 4;   // 32x32 tiles, 4 waves/block
    const int blkF1  = ((MTOK / 32) * (DFF / 32)) / 4;
    const int blkF2  = ((MTOK / 32) * (128 / 32)) / 4;
    const int blkWo  = ((MTOK / 16) * (DMODEL / 16)) / 4;   // 16x16 TDM variant

    for (int l = 0; l < NLAYER; ++l) {
        gemm_bf16_wmma<MODE_BF16><<<blkD, 128, 0, stream>>>(
            xB, DMODEL, WqT + (size_t)l * eD, bq + l * DMODEL,
            MTOK, DMODEL, DMODEL, nullptr, nullptr, 0, qB, DMODEL);
        gemm_bf16_wmma<MODE_BF16><<<blkD, 128, 0, stream>>>(
            xB, DMODEL, WkT + (size_t)l * eD, bk + l * DMODEL,
            MTOK, DMODEL, DMODEL, nullptr, nullptr, 0, kB, DMODEL);
        gemm_bf16_wmma<MODE_VT><<<blkD, 128, 0, stream>>>(
            xvB, DMODEL, WvT + (size_t)l * eD, bv + l * DMODEL,
            MTOK, DMODEL, DMODEL, nullptr, nullptr, 0, vtB, 0);

        if (l & 1)
            attn_kernel<1><<<BATCH * NHEAD * (NSEQ / 16), 32, 0, stream>>>(qB, kB, vtB, mask, oB);
        else
            attn_kernel<0><<<BATCH * NHEAD * (NSEQ / 16), 32, 0, stream>>>(qB, kB, vtB, mask, oB);

        gemm_wo_tdm<<<blkWo, 128, 0, stream>>>(
            oB, WoT + (size_t)l * eD, bo + l * DMODEL, tmp, xF);
        layernorm_kernel<<<MTOK / 4, 128, 0, stream>>>(
            tmp, g1 + l * DMODEL, be1 + l * DMODEL, xF, xB);

        gemm_bf16_wmma<MODE_GELU><<<blkF1, 128, 0, stream>>>(
            xB, DMODEL, W1T + (size_t)l * e1, b1 + l * DFF,
            MTOK, DFF, DMODEL, nullptr, nullptr, 0, h1B, DFF);
        for (int g = 0; g < 2; ++g) {
            gemm_bf16_wmma<MODE_F32R><<<blkF2, 128, 0, stream>>>(
                h1B + g * 512, DFF, W2T + (size_t)l * e2, b2 + l * 128,
                MTOK, 128, 512, tmp + g * 128, xF + g * 128, DMODEL, nullptr, 0);
        }
        layernorm_kernel<<<MTOK / 4, 128, 0, stream>>>(
            tmp, g2 + l * DMODEL, be2 + l * DMODEL, xF, xB);
    }

    // final attention_norm -> d_out (fp32)
    layernorm_kernel<<<MTOK / 4, 128, 0, stream>>>(xF, gf, bf_, (float*)d_out, nullptr);
}
